// attention_layer_78297253806160
// MI455X (gfx1250) — compile-verified, matrix operand order checked by reading, not run on hardware
//
#include <hip/hip_runtime.h>
#include <stdint.h>

typedef __attribute__((ext_vector_type(16))) __bf16 v16bf;
typedef __attribute__((ext_vector_type(8)))  float  v8f;
typedef uint16_t bf16_t;
typedef uint32_t u32;

#define B_    16
#define N_    512
#define LOW_  1024
#define HIGH_ 4096
#define BIGV  9e15f

// round-to-nearest-even f32 -> bf16 (bit pattern)
__device__ __forceinline__ bf16_t f2bf(float f) {
  u32 u = __builtin_bit_cast(u32, f);
  u32 r = u + 0x7FFFu + ((u >> 16) & 1u);
  return (bf16_t)(r >> 16);
}

__global__ __launch_bounds__(256) void f32_to_bf16_k(const float* __restrict__ in,
                                                     bf16_t* __restrict__ out, int n) {
  int i = blockIdx.x * blockDim.x + threadIdx.x;
  if (i < n) out[i] = f2bf(in[i]);
}

// f32 [R][C] row-major -> bf16 [C][R] (transpose + convert), 64x64 LDS tiles.
__global__ __launch_bounds__(256) void cvt_transpose_k(const float* __restrict__ in,
                                                       bf16_t* __restrict__ out,
                                                       int R, int C) {
  __shared__ bf16_t t[64][65];
  const int c0 = blockIdx.x * 64, r0 = blockIdx.y * 64;
  const int tx = threadIdx.x & 63, ty = threadIdx.x >> 6;  // ty: 0..3
#pragma unroll
  for (int i = 0; i < 64; i += 4)
    t[ty + i][tx] = f2bf(in[(size_t)(r0 + ty + i) * C + c0 + tx]);
  __syncthreads();
#pragma unroll
  for (int i = 0; i < 64; i += 4)
    out[(size_t)(c0 + ty + i) * R + r0 + tx] = t[tx][ty + i];
}

// bf16 [R][C] -> bf16 [C][R], batched via blockIdx.z (contiguous R*C per batch).
__global__ __launch_bounds__(256) void transpose_bf16_k(const bf16_t* __restrict__ in,
                                                        bf16_t* __restrict__ out,
                                                        int R, int C) {
  __shared__ bf16_t t[64][65];
  const bf16_t* ib = in + (size_t)blockIdx.z * R * C;
  bf16_t* ob = out + (size_t)blockIdx.z * R * C;
  const int c0 = blockIdx.x * 64, r0 = blockIdx.y * 64;
  const int tx = threadIdx.x & 63, ty = threadIdx.x >> 6;
#pragma unroll
  for (int i = 0; i < 64; i += 4)
    t[ty + i][tx] = ib[(size_t)(r0 + ty + i) * C + c0 + tx];
  __syncthreads();
#pragma unroll
  for (int i = 0; i < 64; i += 4)
    ob[(size_t)(c0 + ty + i) * R + r0 + tx] = t[tx][ty + i];
}

// ---------------------------------------------------------------------------
// WMMA GEMM.  C = alpha * A x B where A is [M][K] (lda) and B is given
// PRE-TRANSPOSED as Bt[n][k] (ldb = k-stride).  LDS tiles: rows of 32 bf16 at
// a 20-dword pitch (order-16 mod 64 -> conflict-free 2x ds_load_b128 fragment
// gathers; 80B rows keep b128 staging 16B-aligned).  Staging is pure straight
// copy -> done with double-buffered GLOBAL_LOAD_ASYNC_TO_LDS_B128 (ASYNCcnt),
// overlapping the DMA of tile s+1 with the 8 WMMAs of tile s.
// ---------------------------------------------------------------------------
__device__ __forceinline__ v16bf load_frag(const u32* lds, int row0, int lane) {
  const int hi = lane >> 4;
  const u32* p = lds + (row0 + (lane & 15)) * 20;
  union { u32 u[8]; v16bf v; } f;
#pragma unroll
  for (int v = 0; v < 8; ++v) {
    int d = (v & 3) + ((v >> 2) << 3) + (hi << 2);  // dwords 0..3 / 8..11 (+4 hi-half)
    f.u[v] = p[d];
  }
  return f.v;
}

// 128 rows x 32 bf16 tile: 256 threads x 2 chunks of 16B, async into LDS.
__device__ __forceinline__ void stage_async(const bf16_t* __restrict__ gp, int ld,
                                            u32* ldsArr, int tid) {
#pragma unroll
  for (int it = 0; it < 2; ++it) {
    int m  = (tid >> 2) + it * 64;
    int kc = (tid & 3) * 8;
    unsigned long long ga = (unsigned long long)(size_t)(gp + (size_t)m * ld + kc);
    unsigned dstoff = (unsigned)(size_t)(const void*)(ldsArr + m * 20 + (kc >> 1));
    asm volatile("global_load_async_to_lds_b128 %0, %1, off"
                 :: "v"(dstoff), "v"(ga) : "memory");
  }
}

template <bool OUTBF16, bool TRANSC>
__global__ __launch_bounds__(256) void gemm_bf16_wmma(
    const bf16_t* __restrict__ A, const bf16_t* __restrict__ Bt,
    void* __restrict__ Cp, int K, int lda, int ldb, int ldc,
    size_t cBatchStride, float alpha) {
  __shared__ u32 lds[4 * 128 * 20];  // [buf0:A,B][buf1:A,B] = 40KB
  const int tid  = threadIdx.x;
  const int lane = tid & 31;
  const int wave = tid >> 5;
  const int wm = wave >> 2;   // 0..1 : 64-row strip
  const int wn = wave & 3;    // 0..3 : 32-col strip
  const int bm = blockIdx.x * 128;
  const int bn = blockIdx.y * 128;

  const bf16_t* Ap = A  + (size_t)bm * lda;
  const bf16_t* Bp = Bt + (size_t)bn * ldb;

  v8f acc[4][2];
#pragma unroll
  for (int i = 0; i < 4; ++i)
#pragma unroll
    for (int j = 0; j < 2; ++j)
#pragma unroll
      for (int r = 0; r < 8; ++r) acc[i][j][r] = 0.f;

  const int nsteps = K >> 5;
  // prologue: kick off tile 0 into buffer 0
  stage_async(Ap, lda, lds, tid);
  stage_async(Bp, ldb, lds + 2560, tid);

  for (int s = 0; s < nsteps; ++s) {
    // buffer offsets via integer select (no LDS-pointer arrays: they force an
    // addrspacecast static initializer that ld.lld rejects)
    const int coff = (s & 1) ? 2 * 2560 : 0;
    const int noff = 2 * 2560 - coff;
    u32* cA = lds + coff;
    u32* cB = lds + coff + 2560;
    // all waves done computing on the other buffer (tile s-1) before DMA overwrite
    __syncthreads();
    const bool more = (s + 1) < nsteps;
    if (more) {
      stage_async(Ap + (s + 1) * 32, lda, lds + noff, tid);
      stage_async(Bp + (s + 1) * 32, ldb, lds + noff + 2560, tid);
      asm volatile("s_wait_asynccnt 0x4" ::: "memory");  // tile s landed; s+1 in flight
    } else {
      asm volatile("s_wait_asynccnt 0x0" ::: "memory");
    }
    __syncthreads();  // every wave's copies for tile s visible

    v16bf bf0 = load_frag(cB, wn * 32,      lane);
    v16bf bf1 = load_frag(cB, wn * 32 + 16, lane);
#pragma unroll
    for (int i = 0; i < 4; ++i) {
      v16bf af = load_frag(cA, wm * 64 + i * 16, lane);
      acc[i][0] = __builtin_amdgcn_wmma_f32_16x16x32_bf16(
          false, af, false, bf0, (short)0, acc[i][0], false, false);
      acc[i][1] = __builtin_amdgcn_wmma_f32_16x16x32_bf16(
          false, af, false, bf1, (short)0, acc[i][1], false, false);
    }
  }

  // ---- epilogue ----
  // C/D layout: VGPR r -> row = r + (lane>=16 ? 8 : 0), col = lane&15
  const int hi  = lane >> 4;
  const int col = lane & 15;
#pragma unroll
  for (int i = 0; i < 4; ++i)
#pragma unroll
    for (int j = 0; j < 2; ++j) {
      int rbase = bm + wm * 64 + i * 16 + hi * 8;
      int cbase = bn + wn * 32 + j * 16 + col;
      if (TRANSC) {
        // transposed bf16 output, batched over 512-row groups:
        // C[(m>>9)*cBatchStride + col*ldc + (m&511)] ; lane's 8 rows contiguous
        union { uint4 q; bf16_t h[8]; } pk;
#pragma unroll
        for (int r = 0; r < 8; ++r) pk.h[r] = f2bf(acc[i][j][r] * alpha);
        size_t idx = (size_t)(rbase >> 9) * cBatchStride +
                     (size_t)cbase * ldc + (rbase & 511);
        *(uint4*)((bf16_t*)Cp + idx) = pk.q;
      } else {
#pragma unroll
        for (int r = 0; r < 8; ++r) {
          float v = acc[i][j][r] * alpha;
          if (OUTBF16)
            ((bf16_t*)Cp)[(size_t)(rbase + r) * ldc + cbase] = f2bf(v);
          else
            ((float*)Cp)[(size_t)(rbase + r) * ldc + cbase] = v;
        }
      }
    }
}

// One block per row l: masked min-max normalize + softmax over H=4096.
__global__ __launch_bounds__(256) void minmax_softmax_k(
    const float* __restrict__ e, const int* __restrict__ linkage,
    bf16_t* __restrict__ attn) {
  const int l = blockIdx.x;
  const int tid = threadIdx.x;
  __shared__ float ebuf[HIGH_];
  __shared__ unsigned char mbuf[HIGH_];
  __shared__ float red[256];
  const float* row  = e + (size_t)l * HIGH_;
  const int*   mrow = linkage + (size_t)l * HIGH_;

  float mn = BIGV, mx = -BIGV;
  for (int h = tid; h < HIGH_; h += 256) {
    float v = row[h];
    int m = mrow[h] > 0;
    ebuf[h] = v;
    mbuf[h] = (unsigned char)m;
    mn = fminf(mn, m ? v : BIGV);
    mx = fmaxf(mx, m ? v : -BIGV);
  }
  red[tid] = mn; __syncthreads();
  for (int s = 128; s > 0; s >>= 1) { if (tid < s) red[tid] = fminf(red[tid], red[tid + s]); __syncthreads(); }
  mn = red[0]; __syncthreads();
  red[tid] = mx; __syncthreads();
  for (int s = 128; s > 0; s >>= 1) { if (tid < s) red[tid] = fmaxf(red[tid], red[tid + s]); __syncthreads(); }
  mx = red[0]; __syncthreads();

  float denom = mx - mn;
  if (denom == 0.f) denom = 1e-6f;
  float inv = 1.f / denom;

  float zmax = -BIGV;
  for (int h = tid; h < HIGH_; h += 256) {
    float z = mbuf[h] ? (ebuf[h] - mn) * inv : -BIGV;
    ebuf[h] = z;  // own slots only
    zmax = fmaxf(zmax, z);
  }
  __syncthreads();
  red[tid] = zmax; __syncthreads();
  for (int s = 128; s > 0; s >>= 1) { if (tid < s) red[tid] = fmaxf(red[tid], red[tid + s]); __syncthreads(); }
  zmax = red[0]; __syncthreads();

  float sum = 0.f;
  for (int h = tid; h < HIGH_; h += 256) {
    float ev = __expf(ebuf[h] - zmax);
    ebuf[h] = ev;
    sum += ev;
  }
  __syncthreads();
  red[tid] = sum; __syncthreads();
  for (int s = 128; s > 0; s >>= 1) { if (tid < s) red[tid] += red[tid + s]; __syncthreads(); }
  float rsum = 1.f / red[0];

  bf16_t* orow = attn + (size_t)l * HIGH_;
  for (int h = tid; h < HIGH_; h += 256) orow[h] = f2bf(ebuf[h] * rsum);
}

extern "C" void kernel_launch(void* const* d_in, const int* in_sizes, int n_in,
                              void* d_out, int out_size, void* d_ws, size_t ws_size,
                              hipStream_t stream) {
  (void)in_sizes; (void)n_in; (void)out_size; (void)ws_size;
  const float* att = (const float*)d_in[0];
  const float* wq  = (const float*)d_in[1];
  const float* wk  = (const float*)d_in[2];
  const float* wv  = (const float*)d_in[3];
  const int* linkage = (const int*)d_in[4];
  float* out = (float*)d_out;

  char* p = (char*)d_ws;
  auto carve = [&](size_t bytes) {
    char* r = p;
    p += (bytes + 255) & ~(size_t)255;
    return r;
  };
  bf16_t* attb  = (bf16_t*)carve((size_t)B_ * N_ * LOW_ * 2);   // [8192][1024]
  bf16_t* wqT   = (bf16_t*)carve((size_t)LOW_ * LOW_ * 2);      // [1024][1024] = wq^T
  bf16_t* wkT   = (bf16_t*)carve((size_t)HIGH_ * LOW_ * 2);     // [4096][1024] = wk^T
  bf16_t* wvT   = (bf16_t*)carve((size_t)LOW_ * LOW_ * 2);      // [1024][1024] = wv^T
  bf16_t* QT    = (bf16_t*)carve((size_t)B_ * LOW_ * N_ * 2);   // [B][L][N]  (Q^T)
  bf16_t* KT    = (bf16_t*)carve((size_t)B_ * HIGH_ * N_ * 2);  // [B][H][N]  (K^T)
  bf16_t* Vb    = (bf16_t*)carve((size_t)B_ * N_ * LOW_ * 2);   // [8192][1024]
  float*  ework = (float*)carve((size_t)LOW_ * HIGH_ * 4);      // per-batch reuse
  bf16_t* attnb = (bf16_t*)carve((size_t)LOW_ * HIGH_ * 2);     // per-batch reuse
  bf16_t* attnT = (bf16_t*)carve((size_t)HIGH_ * LOW_ * 2);     // per-batch reuse

  // conversions (+ weight transposes so GEMM B-tiles are straight copies)
  f32_to_bf16_k<<<(B_ * N_ * LOW_ + 255) / 256, 256, 0, stream>>>(att, attb, B_ * N_ * LOW_);
  cvt_transpose_k<<<dim3(LOW_ / 64, LOW_ / 64), 256, 0, stream>>>(wq, wqT, LOW_, LOW_);
  cvt_transpose_k<<<dim3(HIGH_ / 64, LOW_ / 64), 256, 0, stream>>>(wk, wkT, LOW_, HIGH_);
  cvt_transpose_k<<<dim3(LOW_ / 64, LOW_ / 64), 256, 0, stream>>>(wv, wvT, LOW_, LOW_);

  const int M = B_ * N_;  // 8192
  // Q^T = (att @ wq)^T  -> QT [B][L][N]
  gemm_bf16_wmma<true, true><<<dim3(M / 128, LOW_ / 128), 256, 0, stream>>>(
      attb, wqT, QT, LOW_, LOW_, LOW_, N_, (size_t)LOW_ * N_, 1.f);
  // K^T = (att @ wk)^T  -> KT [B][H][N]
  gemm_bf16_wmma<true, true><<<dim3(M / 128, HIGH_ / 128), 256, 0, stream>>>(
      attb, wkT, KT, LOW_, LOW_, LOW_, N_, (size_t)HIGH_ * N_, 1.f);
  // V = att @ wv  -> Vb [8192][1024] (natural: V is the A-matrix later)
  gemm_bf16_wmma<true, false><<<dim3(M / 128, LOW_ / 128), 256, 0, stream>>>(
      attb, wvT, Vb, LOW_, LOW_, LOW_, LOW_, 0, 1.f);

  const float rsqrt_dk = 1.f / 64.f;  // 1/sqrt(4096)
  for (int b = 0; b < B_; ++b) {
    const bf16_t* QTb = QT + (size_t)b * LOW_ * N_;
    const bf16_t* KTb = KT + (size_t)b * HIGH_ * N_;
    const bf16_t* Vbb = Vb + (size_t)b * N_ * LOW_;
    float* outb = out + (size_t)b * N_ * HIGH_;
    // e = (Q^T K)/sqrt(dk) : A = QT_b [L][N] (lda=N_), Bt = KT_b [H][N] (ldb=N_)
    gemm_bf16_wmma<false, false><<<dim3(LOW_ / 128, HIGH_ / 128), 256, 0, stream>>>(
        QTb, KTb, ework, N_, N_, N_, HIGH_, 0, rsqrt_dk);
    minmax_softmax_k<<<LOW_, 256, 0, stream>>>(ework, linkage, attnb);
    // attn^T for the x GEMM's B-tile straight copies
    transpose_bf16_k<<<dim3(HIGH_ / 64, LOW_ / 64, 1), 256, 0, stream>>>(attnb, attnT, LOW_, HIGH_);
    // x[b] = V[b] @ attn : A = Vb [N][L] (lda=L), Bt = attnT [H][L] (ldb=L)
    gemm_bf16_wmma<false, false><<<dim3(N_ / 128, HIGH_ / 128), 256, 0, stream>>>(
        Vbb, attnT, outb, LOW_, LOW_, LOW_, HIGH_, 0, 1.f);
  }
}
